// TrainOT_50130858279693
// MI455X (gfx1250) — matrix operand order repaired and checked by reading, not compiled, not run on hardware
//
#include <hip/hip_runtime.h>
#include <hip/hip_bf16.h>
#include <math.h>

// Problem constants (match reference)
#define NN      8192
#define M_EGO   6144
#define D_IN    3
#define GCN_W   64
#define REL_W   64
#define TRAIN_W 256
#define OUT_W   3
#define OUTCOLS (NN * OUT_W)   // 24576
#define DEG_CHUNKS 48          // 6144 / 128
#define SRED_CHUNKS 48         // 6144 / 128
#define KSPLIT  2
#define KSEG    (M_EGO / KSPLIT)   // 3072

typedef __attribute__((ext_vector_type(16))) _Float16 v16h;
typedef __attribute__((ext_vector_type(8)))  float    v8f;

// ---------------------------------------------------------------------------
// WMMA operand slot maps (wave32, 16-bit data), per CDNA5 ISA 7.12.2.
// A-matrix 16x32 (MxK): lane = m + 16*(k_in_16 >= 8); VGPR v holds K pairs
//   {0..7,16..23} for lanes<16 and {8..15,24..31} for lanes>=16.
// B-matrix 32x16 (KxN): same K slot order as A: lane = n + 16*(k_in_16 >= 8).
// LDS is laid out so a fragment is 16 contiguous halves at lane*16.
// ---------------------------------------------------------------------------
__device__ __forceinline__ int slotA(int m, int k) {
    int kin  = k & 15;
    int hs   = (kin >> 3) & 1;
    int v    = ((k & 16) ? 4 : 0) + ((kin & 7) >> 1);
    int e    = (v << 1) + (k & 1);
    return (m + (hs << 4)) * 16 + e;
}
__device__ __forceinline__ int slotB(int n, int k) {
    return (n + (((k >> 3) & 1) << 4)) * 16 + ((k & 16) ? 8 : 0) + (k & 7);
}

// ---------------------------------------------------------------------------
// Deterministic degree: stage 1 writes per-chunk partial column sums.
// part[b][i] = sum_{j in chunk b} adj[t, ego[j], ego[i]]
// ---------------------------------------------------------------------------
__global__ __launch_bounds__(256)
void degpart_kernel(const float* __restrict__ adj, const int* __restrict__ ego,
                    const int* __restrict__ tstep, float* __restrict__ part) {
    int i  = blockIdx.x * 256 + threadIdx.x;
    int b  = blockIdx.y;
    int j0 = b * 128;
    size_t base = (size_t)(*tstep) * NN * NN;
    int egoi = ego[i];
    float s = 0.0f;
    for (int jj = 0; jj < 128; ++jj)
        s += adj[base + (size_t)ego[j0 + jj] * NN + egoi];
    part[(size_t)b * M_EGO + i] = s;
}

// Stage 2: dis[i] = rsqrt(1 + sum_b part[b][i])   (fixed summation order)
__global__ void dis_kernel(const float* __restrict__ part, float* __restrict__ dis) {
    int i = blockIdx.x * blockDim.x + threadIdx.x;
    if (i >= M_EGO) return;
    float d = 1.0f;                    // self loop
    for (int b = 0; b < DEG_CHUNKS; ++b)
        d += part[(size_t)b * M_EGO + i];
    dis[i] = rsqrtf(d);
}

// Xs1[i,c] = dis[i] * (feats[i,:] @ W1[:,c])   (bias applied after propagation)
__global__ void xs1_kernel(const float* __restrict__ pos, const int* __restrict__ ego,
                           const int* __restrict__ tstep, const float* __restrict__ W1,
                           const float* __restrict__ dis, _Float16* __restrict__ xs1) {
    int id = blockIdx.x * blockDim.x + threadIdx.x;   // M*64 threads
    int i = id >> 6, c = id & 63;
    size_t pb = ((size_t)(*tstep) * NN + (size_t)ego[i]) * D_IN;
    float acc = pos[pb + 0] * W1[0 * GCN_W + c]
              + pos[pb + 1] * W1[1 * GCN_W + c]
              + pos[pb + 2] * W1[2 * GCN_W + c];
    xs1[id] = (_Float16)(dis[i] * acc);
}

// Xs2[i,c] = dis[i] * (h[i,:] @ W2[:,c])
__global__ void xs2_kernel(const float* __restrict__ h, const float* __restrict__ W2,
                           const float* __restrict__ dis, _Float16* __restrict__ xs2) {
    int id = blockIdx.x * blockDim.x + threadIdx.x;
    int i = id >> 6, c = id & 63;
    float acc = 0.0f;
    for (int k = 0; k < GCN_W; ++k)
        acc += h[(size_t)i * GCN_W + k] * W2[k * GCN_W + c];
    xs2[id] = (_Float16)(dis[i] * acc);
}

// ---------------------------------------------------------------------------
// Propagation GEMM, K-split in two deterministic halves (blockIdx.y):
//   partial[s][i,c] = sum_{j in K-seg s} adj[t,ego_j,ego_i] * Xs[j,c]
// Block = 128 threads (4 waves), 16-row tile, full 64 cols, K-steps of 32.
// Software-pipelined with double-buffered LDS; the K-segment's ego indices
// live in LDS so the gather has no global->global dependent chain. One
// barrier per K-step; one v_wmma_f32_16x16x32_f16 per wave per K-step.
// ---------------------------------------------------------------------------
__global__ __launch_bounds__(128)
void gcn_spmm_wmma(const float* __restrict__ adj, const int* __restrict__ ego,
                   const int* __restrict__ tstep, const _Float16* __restrict__ xs,
                   float* __restrict__ partial) {
    __shared__ int egoJ[KSEG];
    __shared__ alignas(32) _Float16 swzA[2][32 * 16];
    __shared__ alignas(32) _Float16 swzB[2][4][32 * 16];

    const int tid = threadIdx.x;
    const int i0  = blockIdx.x * 16;
    const int js  = blockIdx.y * KSEG;          // K-segment base (global j)
    const size_t adjbase = (size_t)(*tstep) * NN * NN;

    // cooperative copy of this segment's ego indices into LDS (vectorized)
    #pragma unroll
    for (int e = tid; e < KSEG / 4; e += 128)
        reinterpret_cast<int4*>(egoJ)[e] =
            reinterpret_cast<const int4*>(ego + js)[e];

    // per-thread invariants: A tile element (ii, k) with e = tid + 128u
    // => ii = tid & 15 (constant), k = (tid >> 4) + 8u
    const int aii   = tid & 15;
    const int akb   = tid >> 4;                 // k base, rows akb + 8u
    const int acol  = ego[i0 + aii];            // adjacency column (one load)
    const int bk    = tid >> 2;                 // B staging: K row 0..31
    const int bc0   = (tid & 3) * 16;           // B staging: col base
    const int bgrp  = tid & 3;                  // B staging: col group

    __syncthreads();

    float aReg[4];
    v16h  bReg;

    // ---- prologue: fetch tile jl = 0 into registers
    #pragma unroll
    for (int u = 0; u < 4; ++u)
        aReg[u] = adj[adjbase + (size_t)egoJ[akb + 8 * u] * NN + acol];
    bReg = *reinterpret_cast<const v16h*>(&xs[(size_t)(js + bk) * GCN_W + bc0]);

    v8f acc = {};
    int buf = 0;
    for (int jl = 0; jl < KSEG; jl += 32) {
        // ---- commit staged registers to LDS[buf] in WMMA operand layout
        #pragma unroll
        for (int u = 0; u < 4; ++u)
            swzA[buf][slotA(aii, akb + 8 * u)] = (_Float16)aReg[u]; // exact 0/1
        #pragma unroll
        for (int u = 0; u < 16; ++u)
            swzB[buf][bgrp][slotB(u, bk)] = bReg[u];
        __syncthreads();

        // ---- issue next tile's loads (LDS index -> global gather, no
        //      global->global dependency); overlaps the WMMA below
        int jn = jl + 32;
        if (jn < KSEG) {
            #pragma unroll
            for (int u = 0; u < 4; ++u)
                aReg[u] = adj[adjbase + (size_t)egoJ[jn + akb + 8 * u] * NN + acol];
            bReg = *reinterpret_cast<const v16h*>(
                       &xs[(size_t)(js + jn + bk) * GCN_W + bc0]);
        }

        // ---- fragments + matrix op
        int lane = tid & 31;
        int w    = tid >> 5;
        v16h a = *reinterpret_cast<const v16h*>(&swzA[buf][lane * 16]);
        v16h b = *reinterpret_cast<const v16h*>(&swzB[buf][w][lane * 16]);
        acc = __builtin_amdgcn_wmma_f32_16x16x32_f16(
                  false, a, false, b, (short)0, acc, false, false);
        buf ^= 1;
    }

    // ---- write raw partial sums (D layout: VGPR r -> row r / r+8)
    int lane = tid & 31;
    int w    = tid >> 5;
    int hs   = lane >> 4;
    int col  = w * 16 + (lane & 15);
    float* pout = partial + (size_t)blockIdx.y * (M_EGO * GCN_W);
    #pragma unroll
    for (int r = 0; r < 8; ++r) {
        int i = i0 + r + hs * 8;
        pout[(size_t)i * GCN_W + col] = acc[r];
    }
}

// Combine K-split partials, apply normalization/self-loop/bias (+relu):
// Y[i,c] = dis[i]*(p0[i,c] + p1[i,c] + Xs[i,c]) + bias[c]
__global__ void combine_kernel(const float* __restrict__ partial,
                               const _Float16* __restrict__ xs,
                               const float* __restrict__ dis,
                               const float* __restrict__ bias,
                               float* __restrict__ outY, int do_relu) {
    int id = blockIdx.x * blockDim.x + threadIdx.x;   // M*64 threads
    int i = id >> 6, c = id & 63;
    float s = partial[id] + partial[(size_t)(M_EGO * GCN_W) + id] + (float)xs[id];
    float y = dis[i] * s + bias[c];
    if (do_relu) y = fmaxf(y, 0.0f);
    outY[id] = y;
}

// ---------------------------------------------------------------------------
// kvec = trainOT@Wk + bk ; wqk = Wq@kvec ; bqk = bq.kvec   (one block, 64 thr)
// ---------------------------------------------------------------------------
__global__ void kvec_kernel(const float* __restrict__ trainOT, const float* __restrict__ Wk,
                            const float* __restrict__ bk, const float* __restrict__ Wq,
                            const float* __restrict__ bq, float* __restrict__ kw) {
    __shared__ float kv[REL_W];
    int c = threadIdx.x;                         // 64 threads
    float acc = bk[c];
    for (int k = 0; k < TRAIN_W; ++k) acc += trainOT[k] * Wk[k * REL_W + c];
    kv[c] = acc;
    __syncthreads();
    float wv = 0.0f;
    for (int j = 0; j < REL_W; ++j) wv += Wq[c * REL_W + j] * kv[j];
    kw[c] = wv;                                  // wqk
    if (c == 0) {
        float b = 0.0f;
        for (int j = 0; j < REL_W; ++j) b += bq[j] * kv[j];
        kw[64] = b;                              // bqk
    }
}

// rel[i] = sigmoid((g[i,:].wqk + bqk) / sqrt(64))
__global__ void rel_kernel(const float* __restrict__ g, const float* __restrict__ kw,
                           float* __restrict__ rel) {
    int i = blockIdx.x * blockDim.x + threadIdx.x;
    if (i >= M_EGO) return;
    float acc = kw[64];
    for (int c = 0; c < GCN_W; ++c) acc += g[(size_t)i * GCN_W + c] * kw[c];
    acc *= 0.125f;                               // 1/sqrt(REL)
    rel[i] = 1.0f / (1.0f + expf(-acc));
}

// Deterministic weighted reduction, stage 1 (per 128-row chunk):
// spart[b][c<64] = sum_i rel[i]*g[i,c] ; spart[b][64] = sum_i rel[i]
__global__ __launch_bounds__(96)
void sredpart_kernel(const float* __restrict__ g, const float* __restrict__ rel,
                     float* __restrict__ spart) {
    int c  = threadIdx.x;
    int r0 = blockIdx.x * 128;
    if (c < GCN_W) {
        float s = 0.0f;
        for (int r = 0; r < 128; ++r)
            s += rel[r0 + r] * g[(size_t)(r0 + r) * GCN_W + c];
        spart[(size_t)blockIdx.x * 72 + c] = s;
    } else if (c == GCN_W) {
        float sr = 0.0f;
        for (int r = 0; r < 128; ++r) sr += rel[r0 + r];
        spart[(size_t)blockIdx.x * 72 + 64] = sr;
    }
}

// Stage 2 (fixed order) + new_trainOT[c] = trainOT[c] + s@Wv[:,c] + srel*bv[c]
__global__ __launch_bounds__(256)
void nt_kernel(const float* __restrict__ trainOT, const float* __restrict__ Wv,
               const float* __restrict__ bv, const float* __restrict__ spart,
               float* __restrict__ nt_ws, float* __restrict__ nt_out) {
    __shared__ float sacc[72];
    int c = threadIdx.x;                          // 256 threads
    if (c < 65) {
        float s = 0.0f;
        for (int b = 0; b < SRED_CHUNKS; ++b) s += spart[(size_t)b * 72 + c];
        sacc[c] = s;
    }
    __syncthreads();
    float acc = trainOT[c] + sacc[64] * bv[c];
    for (int k = 0; k < GCN_W; ++k) acc += sacc[k] * Wv[k * TRAIN_W + c];
    nt_ws[c]  = acc;
    nt_out[c] = acc;
}

// out[col] = new_trainOT @ Ws[:,col] + bs[col]
__global__ void out_kernel(const float* __restrict__ nt, const float* __restrict__ Ws,
                           const float* __restrict__ bs, float* __restrict__ out) {
    int col = blockIdx.x * blockDim.x + threadIdx.x;   // 24576
    float acc = bs[col];
    for (int k = 0; k < TRAIN_W; ++k) acc += nt[k] * Ws[(size_t)k * OUTCOLS + col];
    out[col] = acc;
}

// ---------------------------------------------------------------------------
extern "C" void kernel_launch(void* const* d_in, const int* in_sizes, int n_in,
                              void* d_out, int out_size, void* d_ws, size_t ws_size,
                              hipStream_t stream) {
    const float* pos   = (const float*)d_in[0];
    const float* adj   = (const float*)d_in[1];
    const int*   ego   = (const int*)  d_in[2];
    const int*   tstep = (const int*)  d_in[3];
    const float* W1    = (const float*)d_in[4];
    const float* b1    = (const float*)d_in[5];
    const float* W2    = (const float*)d_in[6];
    const float* b2    = (const float*)d_in[7];
    const float* trOT  = (const float*)d_in[8];
    const float* Wq    = (const float*)d_in[9];
    const float* bq    = (const float*)d_in[10];
    const float* Wk    = (const float*)d_in[11];
    const float* bk    = (const float*)d_in[12];
    const float* Wv    = (const float*)d_in[13];
    const float* bv    = (const float*)d_in[14];
    const float* Ws    = (const float*)d_in[15];
    const float* bs    = (const float*)d_in[16];

    float* out    = (float*)d_out;               // [24576] then new_trainOT [256]
    float* nt_out = out + OUTCOLS;

    // workspace carve-up (32B-aligned, fully rewritten every call)
    float*    degpart = (float*)d_ws;                               // 48*M
    float*    dis     = degpart + (size_t)DEG_CHUNKS * M_EGO;       // M
    _Float16* xs1     = (_Float16*)(dis + M_EGO);                   // M*64 f16
    float*    h       = (float*)(xs1 + (size_t)M_EGO * GCN_W);      // M*64
    _Float16* xs2     = (_Float16*)(h + (size_t)M_EGO * GCN_W);     // M*64 f16
    float*    g       = (float*)(xs2 + (size_t)M_EGO * GCN_W);      // M*64
    float*    kw      = g + (size_t)M_EGO * GCN_W;                  // 65 (+pad 72)
    float*    rel     = kw + 72;                                    // M
    float*    spart   = rel + M_EGO;                                // 48*72
    float*    nt      = spart + (size_t)SRED_CHUNKS * 72;           // 256 (+pad)
    float*    pgem    = nt + 256;                                   // 2*M*64

    const dim3 gemmGrid(M_EGO / 16, KSPLIT);

    degpart_kernel<<<dim3(M_EGO / 256, DEG_CHUNKS), dim3(256), 0, stream>>>(adj, ego, tstep, degpart);
    dis_kernel<<<dim3(M_EGO / 256), dim3(256), 0, stream>>>(degpart, dis);
    xs1_kernel<<<dim3(M_EGO * GCN_W / 256), dim3(256), 0, stream>>>(pos, ego, tstep, W1, dis, xs1);
    gcn_spmm_wmma<<<gemmGrid, dim3(128), 0, stream>>>(adj, ego, tstep, xs1, pgem);
    combine_kernel<<<dim3(M_EGO * GCN_W / 256), dim3(256), 0, stream>>>(pgem, xs1, dis, b1, h, 1);
    xs2_kernel<<<dim3(M_EGO * GCN_W / 256), dim3(256), 0, stream>>>(h, W2, dis, xs2);
    gcn_spmm_wmma<<<gemmGrid, dim3(128), 0, stream>>>(adj, ego, tstep, xs2, pgem);
    combine_kernel<<<dim3(M_EGO * GCN_W / 256), dim3(256), 0, stream>>>(pgem, xs2, dis, b2, g, 0);
    kvec_kernel<<<dim3(1), dim3(REL_W), 0, stream>>>(trOT, Wk, bk, Wq, bq, kw);
    rel_kernel<<<dim3(M_EGO / 256), dim3(256), 0, stream>>>(g, kw, rel);
    sredpart_kernel<<<dim3(M_EGO / 128), dim3(96), 0, stream>>>(g, rel, spart);
    nt_kernel<<<dim3(1), dim3(TRAIN_W), 0, stream>>>(trOT, Wv, bv, spart, nt, nt_out);
    out_kernel<<<dim3(OUTCOLS / 256), dim3(256), 0, stream>>>(nt, Ws, bs, out);
}
